// AttnBlock_65996467470938
// MI455X (gfx1250) — matrix operand order repaired and checked by reading, not compiled, not run on hardware
//
#include <hip/hip_runtime.h>
#include <hip/hip_bf16.h>

typedef __attribute__((ext_vector_type(16))) __bf16 v16bf;
typedef __attribute__((ext_vector_type(8)))  float  v8f;

static constexpr int kB = 8;
static constexpr int kC = 512;
static constexpr int kN = 4096;          // 64*64 spatial positions
static constexpr int kGroups = 32;
static constexpr int kChPerGrp = kC / kGroups;   // 16

// ---------------------------------------------------------------------------
// Weight prep: fp32 -> bf16; wp stored transposed ([i][o]) for proj GEMM B.
// ---------------------------------------------------------------------------
__global__ __launch_bounds__(256) void prep_weights_kernel(
    const float* __restrict__ wq, const float* __restrict__ wk,
    const float* __restrict__ wv, const float* __restrict__ wp,
    __bf16* __restrict__ wqb, __bf16* __restrict__ wkb,
    __bf16* __restrict__ wvb, __bf16* __restrict__ wptb) {
  int i = blockIdx.x * blockDim.x + threadIdx.x;
  if (i >= kC * kC) return;
  wqb[i] = (__bf16)wq[i];
  wkb[i] = (__bf16)wk[i];
  wvb[i] = (__bf16)wv[i];
  int r = i / kC, c = i % kC;
  wptb[c * kC + r] = (__bf16)wp[i];
}

// ---------------------------------------------------------------------------
// GroupNorm: one block per (batch, group). Output bf16 in [b][c][n] layout.
// ---------------------------------------------------------------------------
__global__ __launch_bounds__(256) void groupnorm_kernel(
    const float* __restrict__ x, const float* __restrict__ gw,
    const float* __restrict__ gb, __bf16* __restrict__ hb) {
  const int batch = blockIdx.x >> 5;
  const int grp   = blockIdx.x & 31;
  const int elems = kChPerGrp * kN;  // 65536
  const float* xp = x + ((size_t)batch * kC + (size_t)grp * kChPerGrp) * kN;
  __bf16* hp = hb + ((size_t)batch * kC + (size_t)grp * kChPerGrp) * kN;

  __shared__ float s_sum[256], s_sq[256];
  float sum = 0.f, sq = 0.f;
  for (int i = threadIdx.x; i < elems; i += 256) {
    float v = xp[i];
    sum += v;
    sq  += v * v;
  }
  s_sum[threadIdx.x] = sum;
  s_sq[threadIdx.x]  = sq;
  __syncthreads();
  for (int s = 128; s > 0; s >>= 1) {
    if ((int)threadIdx.x < s) {
      s_sum[threadIdx.x] += s_sum[threadIdx.x + s];
      s_sq[threadIdx.x]  += s_sq[threadIdx.x + s];
    }
    __syncthreads();
  }
  const float mean = s_sum[0] * (1.0f / elems);
  const float var  = s_sq[0] * (1.0f / elems) - mean * mean;
  const float rstd = rsqrtf(var + 1e-6f);

  for (int i = threadIdx.x; i < elems; i += 256) {
    int ch = grp * kChPerGrp + (i >> 12);  // i / kN
    float v = (xp[i] - mean) * rstd * gw[ch] + gb[ch];
    hp[i] = (__bf16)v;
  }
}

// ---------------------------------------------------------------------------
// WMMA GEMM: C[M,N] = A[M,K] * B[K,N], A/B bf16 row-major, fp32 accumulate.
// Block: 256 threads (8 waves). Block tile 128x128, wave tile 32x64 (2 A frags
// x 4 B frags = 8 WMMAs per K-step of 32; each B frag reused twice).
// Double-buffered LDS; B staged in WMMA-fragment order so fragment reads are
// two ds_load_b128 per operand.
// ---------------------------------------------------------------------------

// A fragment: 16x32 bf16. Lane L: M = L&15; K runs [kb,kb+16)
// with kb = 16*(L>>4); element s holds k = kb + s, contiguous in the row-major LDS A tile.
__device__ __forceinline__ v16bf frag_a(const __bf16* ldsA, int mSub) {
  const int lane = threadIdx.x & 31;
  const int m  = lane & 15;
  const int kb = (lane >> 4) << 4;  // 0 or 16
  const __bf16* rowp = ldsA + (mSub + m) * 32;
  v16bf v;
  int4* pv = reinterpret_cast<int4*>(&v);
  pv[0] = *reinterpret_cast<const int4*>(rowp + kb);
  pv[1] = *reinterpret_cast<const int4*>(rowp + kb + 8);
  return v;
}

// B fragment from fragment-ordered LDS: [frag][lane][16 elems].
__device__ __forceinline__ v16bf frag_b(const __bf16* ldsB, int f) {
  const int lane = threadIdx.x & 31;
  const __bf16* p = ldsB + ((f << 5) + lane) * 16;
  v16bf v;
  int4* pv = reinterpret_cast<int4*>(&v);
  pv[0] = *reinterpret_cast<const int4*>(p);
  pv[1] = *reinterpret_cast<const int4*>(p + 8);
  return v;
}

// Epilogue modes
//  0: bf16 transposed store + bias + scale : out[gn*ldOut+gm] = (v+bias[gm])*scale
//  1: bf16 store + bias                    : out[gm*ldOut+gn] = v+bias[gm]
//  3: fp32 store                           : out[gm*ldOut+gn] = v
//  4: bf16 store                           : out[gm*ldOut+gn] = v
//  5: fp32 transposed + bias + residual    : out[gn*ldOut+gm] = res+v+bias[gn]
template <int MODE>
__global__ __launch_bounds__(256) void gemm_kernel(
    const __bf16* __restrict__ A, size_t aStride,
    const __bf16* __restrict__ B, size_t bStride,
    float* __restrict__ outF, __bf16* __restrict__ outB, size_t outStride,
    const float* __restrict__ bias, const float* __restrict__ resid,
    size_t residStride, float scale, int M, int Ncols, int K, int ldOut) {
  // [2] = double buffer
  __shared__ __bf16 ldsA[2][128 * 32];       // row-major A tile (128 rows)
  __shared__ __bf16 ldsB[2][8 * 32 * 16];    // fragment-ordered B tile

  const int bz = blockIdx.z;
  A += (size_t)bz * aStride;
  B += (size_t)bz * bStride;
  if (outF) outF += (size_t)bz * outStride;
  if (outB) outB += (size_t)bz * outStride;
  if (resid) resid += (size_t)bz * residStride;

  const int t  = threadIdx.x;
  const int w  = t >> 5;
  const int wr = w >> 1;               // 0..3 (M sub-tile of 32)
  const int wc = w & 1;                // 0..1 (N sub-tile of 64)
  const int mBase = blockIdx.y * 128;
  const int nBase = blockIdx.x * 128;

  // Staging assignments (per thread)
  const int aRow = t >> 1;             // 0..127
  const int aCol = (t & 1) << 4;       // 0 or 16
  const int bK0  = (t >> 4) << 1;      // even K row: 0,2,...,30
  const int bCg  = (t & 15) << 3;      // N group: 0..120 step 8

  const __bf16* Aptr = A + (size_t)(mBase + aRow) * K + aCol;
  const __bf16* Bptr = B + (size_t)bK0 * Ncols + nBase + bCg;

  auto loadA0 = [&](int kBase) {
    return *reinterpret_cast<const int4*>(Aptr + kBase);
  };
  auto loadA1 = [&](int kBase) {
    return *reinterpret_cast<const int4*>(Aptr + kBase + 8);
  };
  auto loadB0 = [&](int kBase) {
    return *reinterpret_cast<const int4*>(Bptr + (size_t)kBase * Ncols);
  };
  auto loadB1 = [&](int kBase) {
    return *reinterpret_cast<const int4*>(Bptr + (size_t)(kBase + 1) * Ncols);
  };
  auto storeTiles = [&](int buf, const int4& ra0, const int4& ra1,
                        const int4& rb0, const int4& rb1) {
    *reinterpret_cast<int4*>(&ldsA[buf][aRow * 32 + aCol]) = ra0;
    *reinterpret_cast<int4*>(&ldsA[buf][aRow * 32 + aCol + 8]) = ra1;
    const unsigned short* r0 = reinterpret_cast<const unsigned short*>(&rb0);
    const unsigned short* r1 = reinterpret_cast<const unsigned short*>(&rb1);
    const int slotBase = (bK0 >> 4) << 4;  // 0 or 16
    const int eOff = bK0 & 15;             // even
#pragma unroll
    for (int i = 0; i < 8; ++i) {
      const int n = bCg + i;
      const int idx = (((n >> 4) << 5) + (slotBase | (n & 15))) * 16 + eOff;
      *reinterpret_cast<unsigned int*>(&ldsB[buf][idx]) =
          (unsigned int)r0[i] | ((unsigned int)r1[i] << 16);
    }
  };

  v8f acc[2][4] = {};

  int4 ra0 = loadA0(0);
  int4 ra1 = loadA1(0);
  int4 rb0 = loadB0(0);
  int4 rb1 = loadB1(0);
  storeTiles(0, ra0, ra1, rb0, rb1);
  int cur = 0;

  for (int kBase = 0; kBase < K; kBase += 32) {
    const bool hasNext = (kBase + 32) < K;
    if (hasNext) {
      ra0 = loadA0(kBase + 32);
      ra1 = loadA1(kBase + 32);
      rb0 = loadB0(kBase + 32);
      rb1 = loadB1(kBase + 32);
      if (kBase + 64 < K) __builtin_prefetch(Aptr + kBase + 64, 0, 1);
    }
    __syncthreads();  // buffer `cur` fully written; prior reads of cur^1 done
    v16bf af0 = frag_a(ldsA[cur], wr * 32);
    v16bf af1 = frag_a(ldsA[cur], wr * 32 + 16);
#pragma unroll
    for (int tt = 0; tt < 4; ++tt) {
      v16bf bf = frag_b(ldsB[cur], wc * 4 + tt);
      acc[0][tt] = __builtin_amdgcn_wmma_f32_16x16x32_bf16(
          false, af0, false, bf, (short)0, acc[0][tt], false, false);
      acc[1][tt] = __builtin_amdgcn_wmma_f32_16x16x32_bf16(
          false, af1, false, bf, (short)0, acc[1][tt], false, false);
    }
    if (hasNext) {
      storeTiles(cur ^ 1, ra0, ra1, rb0, rb1);
      cur ^= 1;
    }
  }

  const int lane  = threadIdx.x & 31;
  const int nn    = lane & 15;
  const int mhalf = (lane >> 4) << 3;  // 0 or 8
#pragma unroll
  for (int sub = 0; sub < 2; ++sub) {
#pragma unroll
    for (int tt = 0; tt < 4; ++tt) {
#pragma unroll
      for (int j = 0; j < 8; ++j) {
        const int gm = mBase + wr * 32 + sub * 16 + mhalf + j;
        const int gn = nBase + wc * 64 + tt * 16 + nn;
        const float v = acc[sub][tt][j];
        if constexpr (MODE == 0) {
          outB[(size_t)gn * ldOut + gm] = (__bf16)((v + bias[gm]) * scale);
        } else if constexpr (MODE == 1) {
          outB[(size_t)gm * ldOut + gn] = (__bf16)(v + bias[gm]);
        } else if constexpr (MODE == 3) {
          outF[(size_t)gm * ldOut + gn] = v;
        } else if constexpr (MODE == 4) {
          outB[(size_t)gm * ldOut + gn] = (__bf16)v;
        } else if constexpr (MODE == 5) {
          const size_t idx = (size_t)gn * ldOut + gm;
          outF[idx] = resid[idx] + v + bias[gn];
        }
      }
    }
  }
}

// ---------------------------------------------------------------------------
// Row softmax: S fp32 [4096][4096] -> P bf16 [4096][4096]
// ---------------------------------------------------------------------------
__global__ __launch_bounds__(256) void softmax_kernel(
    const float* __restrict__ S, __bf16* __restrict__ P) {
  const int row = blockIdx.x;
  const float* sp = S + (size_t)row * kN;
  __bf16* pp = P + (size_t)row * kN;
  __shared__ float red[256];

  float m = -1e30f;
  for (int i = threadIdx.x; i < kN; i += 256) m = fmaxf(m, sp[i]);
  red[threadIdx.x] = m;
  __syncthreads();
  for (int s = 128; s > 0; s >>= 1) {
    if ((int)threadIdx.x < s)
      red[threadIdx.x] = fmaxf(red[threadIdx.x], red[threadIdx.x + s]);
    __syncthreads();
  }
  m = red[0];
  __syncthreads();

  float sum = 0.f;
  for (int i = threadIdx.x; i < kN; i += 256) sum += __expf(sp[i] - m);
  red[threadIdx.x] = sum;
  __syncthreads();
  for (int s = 128; s > 0; s >>= 1) {
    if ((int)threadIdx.x < s) red[threadIdx.x] += red[threadIdx.x + s];
    __syncthreads();
  }
  const float inv = 1.0f / red[0];
  for (int i = threadIdx.x; i < kN; i += 256)
    pp[i] = (__bf16)(__expf(sp[i] - m) * inv);
}

// ---------------------------------------------------------------------------
extern "C" void kernel_launch(void* const* d_in, const int* in_sizes, int n_in,
                              void* d_out, int out_size, void* d_ws,
                              size_t ws_size, hipStream_t stream) {
  const float* x   = (const float*)d_in[0];
  const float* gnw = (const float*)d_in[1];
  const float* gnb = (const float*)d_in[2];
  const float* wq  = (const float*)d_in[3];
  const float* bq  = (const float*)d_in[4];
  const float* wk  = (const float*)d_in[5];
  const float* bk  = (const float*)d_in[6];
  const float* wv  = (const float*)d_in[7];
  const float* bv  = (const float*)d_in[8];
  const float* wp  = (const float*)d_in[9];
  const float* bp  = (const float*)d_in[10];
  float* out = (float*)d_out;

  char* ws = (char*)d_ws;
  size_t off = 0;
  auto alloc = [&](size_t bytes) -> void* {
    void* p = ws + off;
    off = (off + bytes + 255) & ~(size_t)255;
    return p;
  };
  const size_t wsz   = (size_t)kC * kC * sizeof(__bf16);
  const size_t actsz = (size_t)kB * kC * kN * sizeof(__bf16);

  __bf16* wqb  = (__bf16*)alloc(wsz);
  __bf16* wkb  = (__bf16*)alloc(wsz);
  __bf16* wvb  = (__bf16*)alloc(wsz);
  __bf16* wptb = (__bf16*)alloc(wsz);
  __bf16* hb   = (__bf16*)alloc(actsz);                       // [b][c][n]
  __bf16* qt   = (__bf16*)alloc(actsz);                       // [b][n][c] (scaled, +bias)
  __bf16* kb   = (__bf16*)alloc(actsz);                       // [b][c][n]
  __bf16* vt   = (__bf16*)alloc(actsz);                       // [b][n][c]
  __bf16* ot   = (__bf16*)alloc(actsz);                       // [b][n][c]
  float*  S    = (float*)alloc((size_t)kN * kN * sizeof(float));   // per-batch
  __bf16* P    = (__bf16*)alloc((size_t)kN * kN * sizeof(__bf16)); // per-batch

  const size_t actStride = (size_t)kC * kN;  // elements per batch
  const float qscale = 1.0f / sqrtf((float)kC);

  // 1) weights fp32->bf16 (+ wp transpose)
  prep_weights_kernel<<<(kC * kC + 255) / 256, 256, 0, stream>>>(
      wq, wk, wv, wp, wqb, wkb, wvb, wptb);

  // 2) GroupNorm -> hb bf16 [b][c][n]
  groupnorm_kernel<<<kB * kGroups, 256, 0, stream>>>(x, gnw, gnb, hb);

  // 3) Q/K/V projections: D[o,p] = W[o,i] * h[i,p]
  //    Q -> qt [p][o] with bias+scale; K -> kb [o][p] with bias; V -> vt [p][o]
  {
    dim3 grid(kN / 128, kC / 128, kB);
    gemm_kernel<0><<<grid, 256, 0, stream>>>(wqb, 0, hb, actStride, nullptr,
                                             qt, actStride, bq, nullptr, 0,
                                             qscale, kC, kN, kC, kC);
    gemm_kernel<1><<<grid, 256, 0, stream>>>(wkb, 0, hb, actStride, nullptr,
                                             kb, actStride, bk, nullptr, 0,
                                             1.0f, kC, kN, kC, kN);
    gemm_kernel<0><<<grid, 256, 0, stream>>>(wvb, 0, hb, actStride, nullptr,
                                             vt, actStride, bv, nullptr, 0,
                                             1.0f, kC, kN, kC, kC);
  }

  // 4) attention per batch (S/P buffers reused; stream-order serializes)
  for (int b = 0; b < kB; ++b) {
    const __bf16* qtb = qt + (size_t)b * actStride;
    const __bf16* kbb = kb + (size_t)b * actStride;
    const __bf16* vtb = vt + (size_t)b * actStride;
    __bf16* otb = ot + (size_t)b * actStride;

    // scores: S[qp,kp] = qt[qp,:] . k[:,kp]   (scale folded into qt)
    {
      dim3 grid(kN / 128, kN / 128, 1);
      gemm_kernel<3><<<grid, 256, 0, stream>>>(qtb, 0, kbb, 0, S, nullptr, 0,
                                               nullptr, nullptr, 0, 1.0f, kN,
                                               kN, kC, kN);
    }
    // softmax rows -> P bf16
    softmax_kernel<<<kN, 256, 0, stream>>>(S, P);
    // PV: ot[qp,c] = P[qp,:] . vt[:,c]
    {
      dim3 grid(kC / 128, kN / 128, 1);
      gemm_kernel<4><<<grid, 256, 0, stream>>>(P, 0, vtb, 0, nullptr, otb, 0,
                                               nullptr, nullptr, 0, 1.0f, kN,
                                               kC, kN, kC);
    }
  }

  // 5) proj + residual: out[b][o][p] = x[b][o][p] + ot[p,:] . wpT[:,o] + bp[o]
  {
    dim3 grid(kC / 128, kN / 128, kB);
    gemm_kernel<5><<<grid, 256, 0, stream>>>(ot, actStride, wptb, 0, out,
                                             nullptr, actStride, bp, x,
                                             actStride, 1.0f, kN, kC, kC, kN);
  }
}
